// CaterpillarBlock_A2_3_NP5_52493090292263
// MI455X (gfx1250) — matrix operand order repaired and checked by reading, not compiled, not run on hardware
//
#include <hip/hip_runtime.h>
#include <hip/hip_bf16.h>
#include <math.h>

// ---------------- problem constants ----------------
#define Bn   32
#define C_   160
#define C3_  480
#define HH   56
#define WW   56
#define HWs  (HH*WW)          // 3136
#define Pn   (Bn*HWs)         // 100352 pixels, divisible by 64

typedef __attribute__((ext_vector_type(16))) _Float16 v16h;
typedef __attribute__((ext_vector_type(8)))  _Float16 v8h;
typedef __attribute__((ext_vector_type(8)))  float    v8f;

union V16U { v16h v; v8h h[2]; };

__device__ __forceinline__ float gelu_f(float x) {
    return 0.5f * x * (1.0f + erff(x * 0.70710678118654752f));
}

// ---- CDNA5 async global->LDS copy (16B per lane), tracked by ASYNCcnt ----
// Low 32 bits of a generic pointer to __shared__ == wave-relative LDS offset
// (ISA 10.2 aperture rules).
__device__ __forceinline__ void async_b128(void* lds_dst, const void* gsrc) {
    unsigned l = (unsigned)(uintptr_t)lds_dst;
    unsigned long long g = (unsigned long long)(uintptr_t)gsrc;
    asm volatile("global_load_async_to_lds_b128 %0, %1, off"
                 :: "v"(l), "v"(g) : "memory");
}
__device__ __forceinline__ void wait_async0() {
    asm volatile("s_wait_asynccnt 0" ::: "memory");
}

// A fragment: 16x32 f16, rows in LDS (row-major, leading dim ldk).
// lane l: row m = l&15;  e0..7 -> K = kb + (l>>4)*8 + e ; e8..15 -> +16
__device__ __forceinline__ v16h load_a_frag(const _Float16* __restrict__ a,
                                            int ldk, int kb, int lane) {
    int m = lane & 15, hi = lane >> 4;
    const _Float16* p = a + m * ldk + kb + hi * 8;
    V16U t;
    t.h[0] = *(const v8h*)p;
    t.h[1] = *(const v8h*)(p + 16);
    return t.v;
}

// B fragment: 32x16 f16 from weights [N][K] row-major (global/L2).
// lane l: col n = l&15;  e0..7 -> K = kb + (l>>4)*8 + e ; e8..15 -> +16
__device__ __forceinline__ v16h load_b_frag(const _Float16* __restrict__ w,
                                            int K, int kb, int lane) {
    int n = lane & 15, hi = lane >> 4;
    const _Float16* p = w + n * K + kb + hi * 8;
    V16U t;
    t.h[0] = *(const v8h*)p;
    t.h[1] = *(const v8h*)(p + 16);
    return t.v;
}

// ---------------- K0: convert weights to f16, pack concat conv ----------------
__global__ void __launch_bounds__(256)
k_prep(const float* __restrict__ wt, const float* __restrict__ wb,
       const float* __restrict__ wr, const float* __restrict__ wl,
       const float* __restrict__ wc,
       const float* __restrict__ bt, const float* __restrict__ bbs,
       const float* __restrict__ br, const float* __restrict__ bl,
       const float* __restrict__ bc,
       const float* __restrict__ wf1, const float* __restrict__ wf2,
       const float* __restrict__ wfc1, const float* __restrict__ wfc2,
       _Float16* __restrict__ wcat16, float* __restrict__ bcat,
       _Float16* __restrict__ wf1_16, _Float16* __restrict__ wf2_16,
       _Float16* __restrict__ wfc1_16, _Float16* __restrict__ wfc2_16) {
    int tid = blockIdx.x * 256 + threadIdx.x;
    int stride = gridDim.x * 256;
    for (int i = tid; i < C_ * C_; i += stride) {
        int o = i / C_, k = i % C_;
        int g = o >> 5, lo = o & 31;
        const float* w = (g == 0) ? wt : (g == 1) ? wb : (g == 2) ? wr : (g == 3) ? wl : wc;
        wcat16[i] = (_Float16)w[lo * C_ + k];
        wf1_16[i] = (_Float16)wf1[i];
    }
    for (int i = tid; i < C_ * C3_; i += stride) {
        wf2_16[i]  = (_Float16)wf2[i];
        wfc1_16[i] = (_Float16)wfc1[i];
        wfc2_16[i] = (_Float16)wfc2[i];
    }
    for (int i = tid; i < C_; i += stride) {
        int g = i >> 5, lo = i & 31;
        const float* bs = (g == 0) ? bt : (g == 1) ? bbs : (g == 2) ? br : (g == 3) ? bl : bc;
        bcat[i] = bs[lo];
    }
}

// ---------------- K1: h = gelu(bn1(x)), BCHW f32 -> BHWC f16 ----------------
__global__ void __launch_bounds__(256)
k_bn1_gelu(const float* __restrict__ x,
           const float* __restrict__ g, const float* __restrict__ b,
           const float* __restrict__ m, const float* __restrict__ v,
           _Float16* __restrict__ h16) {
    size_t i = (size_t)blockIdx.x * 256 + threadIdx.x;   // over B*C*H*W
    int    c  = (int)((i / HWs) % C_);
    size_t bb = i / ((size_t)C_ * HWs);
    size_t s  = i % HWs;
    float sc  = g[c] * rsqrtf(v[c] + 1e-5f);
    float val = x[i] * sc + (b[c] - m[c] * sc);
    h16[(bb * HWs + s) * C_ + c] = (_Float16)gelu_f(val);
}

// ---------------- K2: y5 = h @ wcat^T + bcat  (M=P, K=160, N=160) ----------------
__global__ void __launch_bounds__(128)
k_gemm_y5(const _Float16* __restrict__ h16, const _Float16* __restrict__ wcat,
          const float* __restrict__ bcat, _Float16* __restrict__ y5) {
    __shared__ _Float16 sA[64 * C_];
    int p0 = blockIdx.x * 64;
    const _Float16* src = h16 + (size_t)p0 * C_;
    for (int i = threadIdx.x; i < (64 * C_) / 8; i += 128)
        async_b128(sA + i * 8, src + i * 8);
    wait_async0();
    __syncthreads();
    int lane = threadIdx.x & 31, wave = threadIdx.x >> 5;
    int mbase = wave * 16, mrow = (lane >> 4) * 8, ncol = lane & 15;
    for (int nt = 0; nt < C_ / 16; ++nt) {
        if (nt + 1 < C_ / 16) __builtin_prefetch(wcat + (size_t)(nt + 1) * 16 * C_, 0, 1);
        v8f acc = {};
        for (int kb = 0; kb < C_; kb += 32) {
            v16h a = load_a_frag(sA + mbase * C_, C_, kb, lane);
            v16h bfr = load_b_frag(wcat + (size_t)(nt * 16) * C_, C_, kb, lane);
            acc = __builtin_amdgcn_wmma_f32_16x16x32_f16(false, a, false, bfr,
                                                         (short)0, acc, false, false);
        }
        int n = nt * 16 + ncol;
        float bias = bcat[n];
        _Float16* out = y5 + (size_t)(p0 + mbase + mrow) * C_ + n;
        #pragma unroll
        for (int r = 0; r < 8; ++r)
            out[(size_t)r * C_] = (_Float16)(acc[r] + bias);
    }
}

// ------- K3: lm = shiftgather(y5) @ wf1^T + bf1 ; g = gelu(bn2(lm)) -------
__global__ void __launch_bounds__(128)
k_gemm_lm(const _Float16* __restrict__ y5, const _Float16* __restrict__ wf1w,
          const float* __restrict__ bf1,
          const float* __restrict__ g2, const float* __restrict__ b2,
          const float* __restrict__ m2, const float* __restrict__ v2,
          _Float16* __restrict__ gout) {
    __shared__ _Float16 sA[64 * C_];
    int p0 = blockIdx.x * 64;          // 64 | 3136 => no batch crossing
    // build shifted concat A: 64 rows x 5 groups of 32 channels (64B chunks)
    for (int i = threadIdx.x; i < 64 * 5; i += 128) {
        int r = i / 5, grp = i % 5;
        int p = p0 + r;
        int s = p % HWs, h = s / WW, w = s % WW;
        int np = p; bool oob = false;
        if      (grp == 0) { if (h + 1 < HH) np = p + WW; else oob = true; }   // t
        else if (grp == 1) { if (h - 1 >= 0) np = p - WW; else oob = true; }   // b
        else if (grp == 2) { if (w - 1 >= 0) np = p - 1;  else oob = true; }   // r
        else if (grp == 3) { if (w + 1 < WW) np = p + 1;  else oob = true; }   // l
        _Float16* d = sA + r * C_ + grp * 32;
        if (oob) {
            uint4 z = {0u, 0u, 0u, 0u};
            ((uint4*)d)[0] = z; ((uint4*)d)[1] = z;
            ((uint4*)d)[2] = z; ((uint4*)d)[3] = z;
        } else {
            const _Float16* sp = y5 + (size_t)np * C_ + grp * 32;
            async_b128(d,      sp);
            async_b128(d + 8,  sp + 8);
            async_b128(d + 16, sp + 16);
            async_b128(d + 24, sp + 24);
        }
    }
    wait_async0();
    __syncthreads();
    int lane = threadIdx.x & 31, wave = threadIdx.x >> 5;
    int mbase = wave * 16, mrow = (lane >> 4) * 8, ncol = lane & 15;
    for (int nt = 0; nt < C_ / 16; ++nt) {
        if (nt + 1 < C_ / 16) __builtin_prefetch(wf1w + (size_t)(nt + 1) * 16 * C_, 0, 1);
        v8f acc = {};
        for (int kb = 0; kb < C_; kb += 32) {
            v16h a = load_a_frag(sA + mbase * C_, C_, kb, lane);
            v16h bfr = load_b_frag(wf1w + (size_t)(nt * 16) * C_, C_, kb, lane);
            acc = __builtin_amdgcn_wmma_f32_16x16x32_f16(false, a, false, bfr,
                                                         (short)0, acc, false, false);
        }
        int n = nt * 16 + ncol;
        float bias = bf1[n];
        float sc = g2[n] * rsqrtf(v2[n] + 1e-5f);
        float of = b2[n] - m2[n] * sc;
        _Float16* out = gout + (size_t)(p0 + mbase + mrow) * C_ + n;
        #pragma unroll
        for (int r = 0; r < 8; ++r) {
            float lm = acc[r] + bias;
            out[(size_t)r * C_] = (_Float16)gelu_f(lm * sc + of);
        }
    }
}

// ---------------- K4a: x_h[b,c,h,v] = sum_w g[b,c,h,w]*wph[v,w] + bph[v] ----------------
__global__ void __launch_bounds__(256)
k_proj_h(const _Float16* __restrict__ g16, const float* __restrict__ wph,
         const float* __restrict__ bph, _Float16* __restrict__ hx) {
    __shared__ _Float16 s[WW * C_];
    int bh = blockIdx.x;                 // b*HH + h
    size_t base = (size_t)bh * WW * C_;
    const uint4* src = (const uint4*)(g16 + base);
    for (int i = threadIdx.x; i < (WW * C_) / 8; i += 256) ((uint4*)s)[i] = src[i];
    __syncthreads();
    for (int oi = threadIdx.x; oi < WW * C_; oi += 256) {
        int vv = oi / C_, c = oi % C_;
        float acc = bph[vv];
        const float* wr = wph + vv * WW;
        #pragma unroll 8
        for (int w = 0; w < WW; ++w) acc += (float)s[w * C_ + c] * wr[w];
        hx[base + (size_t)vv * C_ + c] = (_Float16)acc;
    }
}

// ---------------- K4b: x_w[b,c,u,w] = sum_h g[b,c,h,w]*wpw[u,h] + bpw[u] ----------------
__global__ void __launch_bounds__(256)
k_proj_w(const _Float16* __restrict__ g16, const float* __restrict__ wpw,
         const float* __restrict__ bpw, _Float16* __restrict__ wx) {
    __shared__ _Float16 s[HH * C_];
    int b = blockIdx.x / WW, w = blockIdx.x % WW;
    for (int i = threadIdx.x; i < (HH * C_) / 8; i += 256) {
        int h = i / 20, chunk = i % 20;  // 160 f16 = 20 uint4 per row
        ((uint4*)s)[i] =
            ((const uint4*)(g16 + ((size_t)(b * HH + h) * WW + w) * C_))[chunk];
    }
    __syncthreads();
    for (int oi = threadIdx.x; oi < HH * C_; oi += 256) {
        int u = oi / C_, c = oi % C_;
        float acc = bpw[u];
        const float* wr = wpw + u * HH;
        #pragma unroll 8
        for (int h = 0; h < HH; ++h) acc += (float)s[h * C_ + c] * wr[h];
        wx[((size_t)(b * HH + u) * WW + w) * C_ + c] = (_Float16)acc;
    }
}

// ------- K5: x1 = concat[g,x_h,x_w] @ wf2^T + x   (M=P, K=480, N=160), f32 out -------
__global__ void __launch_bounds__(128)
k_gemm_gm(const _Float16* __restrict__ g16, const _Float16* __restrict__ hx,
          const _Float16* __restrict__ wx, const _Float16* __restrict__ wf2w,
          const float* __restrict__ x, float* __restrict__ x1) {
    __shared__ _Float16 sA[64 * C3_];    // 60 KB
    int p0 = blockIdx.x * 64;
    for (int i = threadIdx.x; i < 64 * 60; i += 128) {   // 60 16B-chunks per row
        int r = i / 60, cchunk = i % 60;
        int seg = cchunk / 20, off = cchunk % 20;
        size_t base = (size_t)(p0 + r) * C_;
        const _Float16* sp = ((seg == 0) ? g16 : (seg == 1) ? hx : wx) + base + off * 8;
        async_b128(sA + (size_t)r * C3_ + cchunk * 8, sp);
    }
    wait_async0();
    __syncthreads();
    int lane = threadIdx.x & 31, wave = threadIdx.x >> 5;
    int mbase = wave * 16, mrow = (lane >> 4) * 8, ncol = lane & 15;
    for (int nt = 0; nt < C_ / 16; ++nt) {
        if (nt + 1 < C_ / 16) __builtin_prefetch(wf2w + (size_t)(nt + 1) * 16 * C3_, 0, 1);
        v8f acc = {};
        for (int kb = 0; kb < C3_; kb += 32) {
            v16h a = load_a_frag(sA + mbase * C3_, C3_, kb, lane);
            v16h bfr = load_b_frag(wf2w + (size_t)(nt * 16) * C3_, C3_, kb, lane);
            acc = __builtin_amdgcn_wmma_f32_16x16x32_f16(false, a, false, bfr,
                                                         (short)0, acc, false, false);
        }
        int n = nt * 16 + ncol;
        #pragma unroll
        for (int r = 0; r < 8; ++r) {
            int p = p0 + mbase + mrow + r;
            int bb = p / HWs, s = p % HWs;
            float val = acc[r] + x[((size_t)bb * C_ + n) * HWs + s];
            x1[(size_t)p * C_ + n] = val;
        }
    }
}

// ---------------- K5b: LayerNorm over C per pixel, wave per pixel ----------------
__global__ void __launch_bounds__(256)
k_ln(const float* __restrict__ x1, const float* __restrict__ lng,
     const float* __restrict__ lnb, _Float16* __restrict__ ln16) {
    int p = blockIdx.x * 8 + (threadIdx.x >> 5);
    int lane = threadIdx.x & 31;
    const float* row = x1 + (size_t)p * C_;
    float s = 0.f, s2 = 0.f;
    #pragma unroll
    for (int i = 0; i < 5; ++i) { float v = row[lane + i * 32]; s += v; s2 += v * v; }
    #pragma unroll
    for (int m = 16; m >= 1; m >>= 1) {
        s  += __shfl_xor(s,  m, 32);
        s2 += __shfl_xor(s2, m, 32);
    }
    float mu   = s  * (1.0f / C_);
    float var  = s2 * (1.0f / C_) - mu * mu;
    float rstd = rsqrtf(var + 1e-5f);
    _Float16* out = ln16 + (size_t)p * C_;
    #pragma unroll
    for (int i = 0; i < 5; ++i) {
        int c = lane + i * 32;
        out[c] = (_Float16)((row[c] - mu) * rstd * lng[c] + lnb[c]);
    }
}

// ---------------- K6: z = gelu(ln @ wfc1^T + bfc1)  (K=160, N=480) ----------------
__global__ void __launch_bounds__(128)
k_gemm_fc1(const _Float16* __restrict__ ln16, const _Float16* __restrict__ wfc1w,
           const float* __restrict__ bfc1, _Float16* __restrict__ z16) {
    __shared__ _Float16 sA[64 * C_];
    int p0 = blockIdx.x * 64;
    const _Float16* src = ln16 + (size_t)p0 * C_;
    for (int i = threadIdx.x; i < (64 * C_) / 8; i += 128)
        async_b128(sA + i * 8, src + i * 8);
    wait_async0();
    __syncthreads();
    int lane = threadIdx.x & 31, wave = threadIdx.x >> 5;
    int mbase = wave * 16, mrow = (lane >> 4) * 8, ncol = lane & 15;
    for (int nt = 0; nt < C3_ / 16; ++nt) {
        if (nt + 1 < C3_ / 16) __builtin_prefetch(wfc1w + (size_t)(nt + 1) * 16 * C_, 0, 1);
        v8f acc = {};
        for (int kb = 0; kb < C_; kb += 32) {
            v16h a = load_a_frag(sA + mbase * C_, C_, kb, lane);
            v16h bfr = load_b_frag(wfc1w + (size_t)(nt * 16) * C_, C_, kb, lane);
            acc = __builtin_amdgcn_wmma_f32_16x16x32_f16(false, a, false, bfr,
                                                         (short)0, acc, false, false);
        }
        int n = nt * 16 + ncol;
        float bias = bfc1[n];
        _Float16* out = z16 + (size_t)(p0 + mbase + mrow) * C3_ + n;
        #pragma unroll
        for (int r = 0; r < 8; ++r)
            out[(size_t)r * C3_] = (_Float16)gelu_f(acc[r] + bias);
    }
}

// ------- K7: out = z @ wfc2^T + bfc2 + x1, stored BCHW f32 (K=480, N=160) -------
__global__ void __launch_bounds__(128)
k_gemm_fc2(const _Float16* __restrict__ z16, const _Float16* __restrict__ wfc2w,
           const float* __restrict__ bfc2, const float* __restrict__ x1,
           float* __restrict__ out) {
    __shared__ _Float16 sA[64 * C3_];    // 60 KB
    int p0 = blockIdx.x * 64;
    const _Float16* src = z16 + (size_t)p0 * C3_;
    for (int i = threadIdx.x; i < (64 * C3_) / 8; i += 128)
        async_b128(sA + i * 8, src + i * 8);
    wait_async0();
    __syncthreads();
    int lane = threadIdx.x & 31, wave = threadIdx.x >> 5;
    int mbase = wave * 16, mrow = (lane >> 4) * 8, ncol = lane & 15;
    for (int nt = 0; nt < C_ / 16; ++nt) {
        if (nt + 1 < C_ / 16) __builtin_prefetch(wfc2w + (size_t)(nt + 1) * 16 * C3_, 0, 1);
        v8f acc = {};
        for (int kb = 0; kb < C3_; kb += 32) {
            v16h a = load_a_frag(sA + mbase * C3_, C3_, kb, lane);
            v16h bfr = load_b_frag(wfc2w + (size_t)(nt * 16) * C3_, C3_, kb, lane);
            acc = __builtin_amdgcn_wmma_f32_16x16x32_f16(false, a, false, bfr,
                                                         (short)0, acc, false, false);
        }
        int n = nt * 16 + ncol;
        float bias = bfc2[n];
        #pragma unroll
        for (int r = 0; r < 8; ++r) {
            int p = p0 + mbase + mrow + r;
            int bb = p / HWs, s = p % HWs;
            float val = acc[r] + bias + x1[(size_t)p * C_ + n];
            out[((size_t)bb * C_ + n) * HWs + s] = val;
        }
    }
}

// ---------------- launch ----------------
extern "C" void kernel_launch(void* const* d_in, const int* in_sizes, int n_in,
                              void* d_out, int out_size, void* d_ws, size_t ws_size,
                              hipStream_t stream) {
    const float* x     = (const float*)d_in[0];
    const float* bn1_g = (const float*)d_in[1];
    const float* bn1_b = (const float*)d_in[2];
    const float* bn1_m = (const float*)d_in[3];
    const float* bn1_v = (const float*)d_in[4];
    const float* wt = (const float*)d_in[5];  const float* bt = (const float*)d_in[6];
    const float* wb = (const float*)d_in[7];  const float* bb = (const float*)d_in[8];
    const float* wr = (const float*)d_in[9];  const float* br = (const float*)d_in[10];
    const float* wl = (const float*)d_in[11]; const float* bl = (const float*)d_in[12];
    const float* wc = (const float*)d_in[13]; const float* bc = (const float*)d_in[14];
    const float* wf1 = (const float*)d_in[15]; const float* bf1 = (const float*)d_in[16];
    const float* bn2_g = (const float*)d_in[17];
    const float* bn2_b = (const float*)d_in[18];
    const float* bn2_m = (const float*)d_in[19];
    const float* bn2_v = (const float*)d_in[20];
    const float* wph = (const float*)d_in[21]; const float* bph = (const float*)d_in[22];
    const float* wpw = (const float*)d_in[23]; const float* bpw = (const float*)d_in[24];
    const float* wf2 = (const float*)d_in[25];
    const float* ln_g = (const float*)d_in[26]; const float* ln_b = (const float*)d_in[27];
    const float* wfc1 = (const float*)d_in[28]; const float* bfc1 = (const float*)d_in[29];
    const float* wfc2 = (const float*)d_in[30]; const float* bfc2 = (const float*)d_in[31];
    float* out = (float*)d_out;

    // workspace carve-up (buffers reused: z overlays h/y5/x_h which are dead by K6)
    char* wsb = (char*)d_ws;
    const size_t PC2  = (size_t)Pn * C_  * 2;   // f16 P x C
    const size_t PC4  = (size_t)Pn * C_  * 4;   // f32 P x C
    const size_t PC32 = (size_t)Pn * C3_ * 2;   // f16 P x 3C
    size_t off = 0;
    auto take = [&](size_t bytes) { size_t o = off; off += (bytes + 255) & ~(size_t)255; return o; };
    size_t zZ    = take(PC32);          // h16 | y5 | hx ; reused as z16
    size_t zWX   = take(PC2);
    size_t zG    = take(PC2);
    size_t zX1   = take(PC4);
    size_t zLN   = take(PC2);
    size_t zWCAT = take((size_t)C_ * C_ * 2);
    size_t zBCAT = take((size_t)C_ * 4);
    size_t zWF1  = take((size_t)C_ * C_ * 2);
    size_t zWF2  = take((size_t)C_ * C3_ * 2);
    size_t zWFC1 = take((size_t)C3_ * C_ * 2);
    size_t zWFC2 = take((size_t)C_ * C3_ * 2);

    _Float16* h16    = (_Float16*)(wsb + zZ);
    _Float16* y5     = (_Float16*)(wsb + zZ + PC2);
    _Float16* hx     = (_Float16*)(wsb + zZ + 2 * PC2);
    _Float16* z16    = (_Float16*)(wsb + zZ);
    _Float16* wx     = (_Float16*)(wsb + zWX);
    _Float16* g16    = (_Float16*)(wsb + zG);
    float*    x1     = (float*)   (wsb + zX1);
    _Float16* ln16   = (_Float16*)(wsb + zLN);
    _Float16* wcat16 = (_Float16*)(wsb + zWCAT);
    float*    bcat   = (float*)   (wsb + zBCAT);
    _Float16* wf1_16 = (_Float16*)(wsb + zWF1);
    _Float16* wf2_16 = (_Float16*)(wsb + zWF2);
    _Float16* wfc1_16= (_Float16*)(wsb + zWFC1);
    _Float16* wfc2_16= (_Float16*)(wsb + zWFC2);

    // K0: weight conversion/packing
    k_prep<<<128, 256, 0, stream>>>(wt, wb, wr, wl, wc, bt, bb, br, bl, bc,
                                    wf1, wf2, wfc1, wfc2,
                                    wcat16, bcat, wf1_16, wf2_16, wfc1_16, wfc2_16);
    // K1: BN1 + GELU + BCHW->BHWC f16
    k_bn1_gelu<<<(Bn * C_ * HWs) / 256, 256, 0, stream>>>(x, bn1_g, bn1_b, bn1_m, bn1_v, h16);
    // K2: five 1x1 convs fused as one 160->160 WMMA GEMM
    k_gemm_y5<<<Pn / 64, 128, 0, stream>>>(h16, wcat16, bcat, y5);
    // K3: shift-gather + wf1 GEMM + BN2 + GELU
    k_gemm_lm<<<Pn / 64, 128, 0, stream>>>(y5, wf1_16, bf1,
                                           bn2_g, bn2_b, bn2_m, bn2_v, g16);
    // K4: spatial mixes
    k_proj_h<<<Bn * HH, 256, 0, stream>>>(g16, wph, bph, hx);
    k_proj_w<<<Bn * WW, 256, 0, stream>>>(g16, wpw, bpw, wx);
    // K5: wf2 GEMM over concat + residual x -> x1 (f32)
    k_gemm_gm<<<Pn / 64, 128, 0, stream>>>(g16, hx, wx, wf2_16, x, x1);
    // K5b: LayerNorm
    k_ln<<<Pn / 8, 256, 0, stream>>>(x1, ln_g, ln_b, ln16);
    // K6: fc1 GEMM + GELU (z16 overlays dead h16/y5/hx)
    k_gemm_fc1<<<Pn / 64, 128, 0, stream>>>(ln16, wfc1_16, bfc1, z16);
    // K7: fc2 GEMM + bias + residual x1, store BCHW f32
    k_gemm_fc2<<<Pn / 64, 128, 0, stream>>>(z16, wfc2_16, bfc2, x1, out);
}
